// GeoSAGEConv_26645977104607
// MI455X (gfx1250) — matrix operand order repaired and checked by reading, not compiled, not run on hardware
//
#include <hip/hip_runtime.h>

// ---------------------------------------------------------------------------
// GeoSAGEConv on MI455X (gfx1250, wave32).
// Dense-adjacency reformulation: segment-mean == row-normalized dense A (200x200).
// Heavy GEMMs run on v_wmma_f32_16x16x32_bf16; fp32 weights are streamed from
// HBM once (LDS double-buffered, bf16-packed) — memory-bound: ~550MB @ 23.3TB/s.
// ---------------------------------------------------------------------------

#define N_NODES   200
#define M_PAD     208          // 13 * 16
#define M_TILES   13
#define N_EDGES   12800
#define NFEAT     16384
#define NHID      4096
#define NCLASS    512

typedef __attribute__((ext_vector_type(16))) __bf16    v16bf;
typedef __attribute__((ext_vector_type(2)))  __bf16    v2bf;
typedef __attribute__((ext_vector_type(8)))  float     v8f;
typedef __attribute__((ext_vector_type(4)))  unsigned  myuint4;
typedef __attribute__((ext_vector_type(4)))  float     myfloat4;

union FragBF {
    myuint4  q[2];
    unsigned u[8];
    v16bf    v;
};

// Native converts: clang lowers these to the gfx1250 bf16 cvt ops
// (v_cvt_pk_bf16_f32 for the pair form) instead of a 4-op manual RNE sequence.
__device__ __forceinline__ unsigned short f2bf(float f) {
    __bf16 h = (__bf16)f;
    return __builtin_bit_cast(unsigned short, h);
}
__device__ __forceinline__ float bf2f(unsigned short b) {
    __bf16 h = __builtin_bit_cast(__bf16, b);
    return (float)h;
}
__device__ __forceinline__ unsigned packbf(float a, float b) {
    v2bf p;
    p.x = (__bf16)a;
    p.y = (__bf16)b;
    return __builtin_bit_cast(unsigned, p);
}

// ---------------------------------------------------------------------------
// 0) zero workspace prefix
// ---------------------------------------------------------------------------
__global__ void zero_kernel(unsigned* __restrict__ p, size_t n) {
    size_t i = (size_t)blockIdx.x * blockDim.x + threadIdx.x;
    size_t stride = (size_t)gridDim.x * blockDim.x;
    for (; i < n; i += stride) p[i] = 0u;
}

// 1) scatter edges into dense adjacency + in-degree counts
__global__ void scatter_kernel(const int* __restrict__ ei,
                               const float* __restrict__ ew,
                               float* __restrict__ Adj,
                               float* __restrict__ cnt) {
    int e = blockIdx.x * blockDim.x + threadIdx.x;
    if (e < N_EDGES) {
        int s = ei[e];               // row 0: source j
        int d = ei[N_EDGES + e];     // row 1: target i
        atomicAdd(&Adj[d * N_NODES + s], ew[e]);
        atomicAdd(&cnt[d], 1.0f);
    }
}

// 2) fold mean-normalization into adjacency rows
__global__ void norm_kernel(float* __restrict__ Adj, const float* __restrict__ cnt) {
    int i = blockIdx.x * blockDim.x + threadIdx.x;
    if (i < N_NODES * N_NODES) {
        int d = i / N_NODES;
        Adj[i] = Adj[i] / fmaxf(cnt[d], 1.0f);
    }
}

// 3) convert x -> bf16 (padded rows pre-zeroed); 2 elements/thread, packed store
__global__ void cvt_kernel(const float* __restrict__ x,
                           unsigned* __restrict__ xb2, size_t npairs) {
    size_t i = (size_t)blockIdx.x * blockDim.x + threadIdx.x;
    if (i < npairs) xb2[i] = packbf(x[2 * i], x[2 * i + 1]);
}

// 4) aggregation: Out_bf16[m,f] = sum_s Ahat[m,s] * In_bf16[s,f]
__global__ void agg_kernel(const float* __restrict__ Adj,
                           const unsigned short* __restrict__ In,
                           unsigned short* __restrict__ Out, int F) {
    int f = blockIdx.x * blockDim.x + threadIdx.x;
    int m = blockIdx.y;
    if (f < F) {
        float acc = 0.0f;
        for (int s = 0; s < N_NODES; ++s)
            acc += Adj[m * N_NODES + s] * bf2f(In[(size_t)s * F + f]);
        Out[(size_t)m * F + f] = f2bf(acc);
    }
}

// ---------------------------------------------------------------------------
// Weight stage: 8 consecutive fp32 -> 4 packed bf16 pairs -> LDS b128 store.
// ---------------------------------------------------------------------------
__device__ __forceinline__ void stage_w(const float* __restrict__ wp,
                                        unsigned* __restrict__ ldst) {
    const myfloat4* w4 = (const myfloat4*)wp;
    myfloat4 w0 = w4[0];
    myfloat4 w1 = w4[1];
    myuint4 o;
    o.x = packbf(w0.x, w0.y);
    o.y = packbf(w0.z, w0.w);
    o.z = packbf(w1.x, w1.y);
    o.w = packbf(w1.z, w1.w);
    *(myuint4*)ldst = o;
    __builtin_prefetch(wp + 256, 0, 1);            // two K-chunks ahead
}

// ---------------------------------------------------------------------------
// One streamed GEMM pass: acc += A[M_PAD x K](bf16) @ W[N x K](fp32)^T tile.
// K chunked by 128: double-buffered LDS (2 x 4KB), 4 WMMAs per chunk,
// one workgroup barrier per chunk.
// ---------------------------------------------------------------------------
__device__ __forceinline__ void gemm_stream(const unsigned short* __restrict__ A,
                                            const float* __restrict__ W,
                                            int K, int ntile,
                                            int t, int lane, int h, int n16, int mrow,
                                            unsigned (&lds)[2][1024],
                                            v8f& acc) {
    const int chunks = K >> 7;                     // K / 128
    // cooperative loader mapping: thread t<256 owns (col nl, 8-float group g)
    const int nl = t >> 4;
    const int g  = t & 15;
    const int so = ((4 * g) >> 4) * 256 + nl * 16 + ((4 * g) & 15); // LDS slot
    const float* wrow = W + (size_t)(ntile * 16 + nl) * K + g * 8;

    const myuint4* arow = (const myuint4*)(A + (size_t)mrow * K);

    if (t < 256) stage_w(wrow, &lds[0][so]);       // preload chunk 0
    __syncthreads();

    for (int c = 0; c < chunks; ++c) {
        const int kk = c << 7;
        if (c + 1 < chunks && t < 256)             // load next chunk (other buffer)
            stage_w(wrow + (size_t)kk + 128, &lds[(c + 1) & 1][so]);

        const unsigned* B = lds[c & 1];
#pragma unroll
        for (int s = 0; s < 4; ++s) {
            const int kb = kk + (s << 5);
            FragBF a, b;
            a.q[0] = arow[(kb + 8 * h) >> 3];          // ISA 7.12.2 A 16x32 layout
            a.q[1] = arow[(kb + 16 + 8 * h) >> 3];
            const myuint4* bls = (const myuint4*)(B + s * 256 + n16 * 16 + 4 * h);
            b.q[0] = bls[0];
            b.q[1] = bls[2];
            acc = __builtin_amdgcn_wmma_f32_16x16x32_bf16(
                false, a.v, false, b.v, (short)0, acc, false, false);
        }
        __syncthreads();                           // next chunk's writes may land
    }
}

// ---------------------------------------------------------------------------
// 5) Fused SAGE GEMM:  D = act( Aa@Wa^T + Ab@Wb^T + bias )
//    Block = 13 waves (416 thr); wave w owns M-tile w; block owns one N=16 strip.
// ---------------------------------------------------------------------------
__global__ __launch_bounds__(M_TILES * 32)
void sage_wmma_kernel(const unsigned short* __restrict__ Aa,
                      const unsigned short* __restrict__ Ab,
                      const float* __restrict__ Wa,
                      const float* __restrict__ Wb,
                      const float* __restrict__ bias,
                      unsigned short* __restrict__ outB,   // bf16 out (or null)
                      float* __restrict__ outF,            // fp32 out (or null)
                      int K, int N, int doRelu) {
    __shared__ unsigned ldsB[2][1024];             // double-buffered weight tile

    const int t    = threadIdx.x;
    const int wave = t >> 5;                       // == M-tile index
    const int lane = t & 31;
    const int h    = lane >> 4;
    const int n16  = lane & 15;
    const int ntile = blockIdx.x;
    const int nglob = ntile * 16 + n16;
    const int mrow  = wave * 16 + n16;

    v8f acc = {0.f, 0.f, 0.f, 0.f, 0.f, 0.f, 0.f, 0.f};

    gemm_stream(Aa, Wa, K, ntile, t, lane, h, n16, mrow, ldsB, acc);
    gemm_stream(Ab, Wb, K, ntile, t, lane, h, n16, mrow, ldsB, acc);

    const float bv = bias[nglob];
#pragma unroll
    for (int r = 0; r < 8; ++r) {
        int m = wave * 16 + r + 8 * h;             // C/D layout (ISA 7.12.2)
        if (m < N_NODES) {
            float val = acc[r] + bv;
            if (doRelu) val = fmaxf(val, 0.0f);
            if (outB) outB[(size_t)m * N + nglob] = f2bf(val);
            if (outF) outF[(size_t)m * N + nglob] = val;
        }
    }
}

// ---------------------------------------------------------------------------
// 6) classifier MLP on Z = H2^T  (512 x 200 -> 100 -> 50 -> 10); tiny
// ---------------------------------------------------------------------------
__global__ void fc_t_kernel(const float* __restrict__ H2,   // [200 x 512]
                            const float* __restrict__ W,    // [OUT x 200]
                            const float* __restrict__ b,
                            float* __restrict__ out, int OUT, int doRelu) {
    int i = blockIdx.x * blockDim.x + threadIdx.x;
    if (i < NCLASS * OUT) {
        int c = i / OUT, j = i % OUT;
        float acc = b[j];
        for (int k = 0; k < N_NODES; ++k)
            acc += H2[k * NCLASS + c] * W[j * N_NODES + k];
        if (doRelu) acc = fmaxf(acc, 0.0f);
        out[c * OUT + j] = acc;
    }
}

__global__ void fc_kernel(const float* __restrict__ in,     // [512 x IN]
                          const float* __restrict__ W,      // [OUT x IN]
                          const float* __restrict__ b,
                          float* __restrict__ out, int IN, int OUT, int doRelu) {
    int i = blockIdx.x * blockDim.x + threadIdx.x;
    if (i < NCLASS * OUT) {
        int c = i / OUT, j = i % OUT;
        float acc = b[j];
        for (int k = 0; k < IN; ++k)
            acc += in[c * IN + k] * W[j * IN + k];
        if (doRelu) acc = fmaxf(acc, 0.0f);
        out[c * OUT + j] = acc;
    }
}

// ---------------------------------------------------------------------------
extern "C" void kernel_launch(void* const* d_in, const int* in_sizes, int n_in,
                              void* d_out, int out_size, void* d_ws, size_t ws_size,
                              hipStream_t stream) {
    (void)in_sizes; (void)n_in; (void)out_size; (void)ws_size;

    const float* x   = (const float*)d_in[0];
    const int*   ei  = (const int*)  d_in[1];
    const float* ew  = (const float*)d_in[2];
    const float* W1l = (const float*)d_in[3];
    const float* W1r = (const float*)d_in[4];
    const float* b1  = (const float*)d_in[5];
    const float* W2l = (const float*)d_in[6];
    const float* W2r = (const float*)d_in[7];
    const float* b2  = (const float*)d_in[8];
    const float* Wc1 = (const float*)d_in[9];
    const float* bc1 = (const float*)d_in[10];
    const float* Wc2 = (const float*)d_in[11];
    const float* bc2 = (const float*)d_in[12];
    const float* Wc3 = (const float*)d_in[13];
    const float* bc3 = (const float*)d_in[14];

    // ---- workspace layout (bytes, 256-aligned slots) ----
    char* ws = (char*)d_ws;
    size_t off = 0;
    auto alloc = [&](size_t bytes) {
        char* p = ws + off;
        off += (bytes + 255) & ~(size_t)255;
        return p;
    };
    float*          Adj   = (float*)alloc((size_t)N_NODES * N_NODES * 4);
    float*          cnt   = (float*)alloc((size_t)N_NODES * 4);
    unsigned short* Xb    = (unsigned short*)alloc((size_t)M_PAD * NFEAT * 2);
    unsigned short* Aggb  = (unsigned short*)alloc((size_t)M_PAD * NFEAT * 2);
    unsigned short* H1b   = (unsigned short*)alloc((size_t)M_PAD * NHID * 2);
    unsigned short* Agg2b = (unsigned short*)alloc((size_t)M_PAD * NHID * 2);
    size_t zero_bytes = off;                       // everything above starts at 0
    float*          H2    = (float*)alloc((size_t)N_NODES * NCLASS * 4);
    float*          Z1    = (float*)alloc((size_t)NCLASS * 100 * 4);
    float*          Z2    = (float*)alloc((size_t)NCLASS * 50 * 4);

    // 0) zero adjacency / counts / padded bf16 buffers
    zero_kernel<<<2048, 256, 0, stream>>>((unsigned*)ws, zero_bytes / 4);

    // 1-2) dense adjacency with mean-normalization folded in
    scatter_kernel<<<(N_EDGES + 255) / 256, 256, 0, stream>>>(ei, ew, Adj, cnt);
    norm_kernel<<<(N_NODES * N_NODES + 255) / 256, 256, 0, stream>>>(Adj, cnt);

    // 3) x -> bf16 (packed pair stores)
    cvt_kernel<<<(N_NODES * NFEAT / 2) / 256, 256, 0, stream>>>(
        x, (unsigned*)Xb, (size_t)N_NODES * NFEAT / 2);

    // 4) agg1 = Ahat @ X
    agg_kernel<<<dim3(NFEAT / 256, N_NODES), 256, 0, stream>>>(Adj, Xb, Aggb, NFEAT);

    // 5) gc1 (the big one): H1 = relu(agg1@W1l^T + X@W1r^T + b1)
    sage_wmma_kernel<<<NHID / 16, M_TILES * 32, 0, stream>>>(
        Aggb, Xb, W1l, W1r, b1, H1b, nullptr, NFEAT, NHID, 1);

    // 6) agg2 = Ahat @ H1
    agg_kernel<<<dim3(NHID / 256, N_NODES), 256, 0, stream>>>(Adj, H1b, Agg2b, NHID);

    // 7) gc2: H2 = agg2@W2l^T + H1@W2r^T + b2   (fp32 out for classifier)
    sage_wmma_kernel<<<NCLASS / 16, M_TILES * 32, 0, stream>>>(
        Agg2b, H1b, W2l, W2r, b2, nullptr, H2, NHID, NCLASS, 0);

    // 8) classifier on Z = H2^T
    fc_t_kernel<<<(NCLASS * 100 + 255) / 256, 256, 0, stream>>>(H2, Wc1, bc1, Z1, 100, 1);
    fc_kernel<<<(NCLASS * 50 + 255) / 256, 256, 0, stream>>>(Z1, Wc2, bc2, Z2, 100, 50, 1);
    fc_kernel<<<(NCLASS * 10 + 255) / 256, 256, 0, stream>>>(Z2, Wc3, bc3, (float*)d_out, 50, 10, 0);
}
